// PretrainedModel_78434692760006
// MI455X (gfx1250) — matrix operand order repaired and checked by reading, not compiled, not run on hardware
//
#include <hip/hip_runtime.h>
#include <hip/hip_bf16.h>

// ---- problem constants (match reference) ----
#define DIM    256
#define MTILE  32          // pairs per tile; 500000 % 32 == 0
#define SD     264         // LDS stride (halves) for d tile, mult of 8 -> 16B-aligned rows
#define SW     264         // LDS stride (halves) for W1^T (512B row + 16B pad)
#define NBLOCKS 2048       // persistent grid

typedef _Float16 v16h __attribute__((ext_vector_type(16)));
typedef _Float16 v8h  __attribute__((ext_vector_type(8)));
typedef _Float16 v4h  __attribute__((ext_vector_type(4)));
typedef float    v8f  __attribute__((ext_vector_type(8)));
typedef float    v4f  __attribute__((ext_vector_type(4)));

typedef unsigned int u32x4 __attribute__((ext_vector_type(4)));
typedef int          i32x4 __attribute__((ext_vector_type(4)));
typedef int          i32x8 __attribute__((ext_vector_type(8)));

union V16 { v16h v; v8h h[2]; };

#if __has_builtin(__builtin_amdgcn_tensor_load_to_lds) && __has_builtin(__builtin_amdgcn_s_wait_tensorcnt)
#define HAVE_TDM 1
#endif

// ---------------------------------------------------------------------------
// Pre-kernel: W1 (f32, [k][n] row-major) -> d_ws as f16 W1^T ([n][k] row-major)
// ---------------------------------------------------------------------------
__global__ __launch_bounds__(256)
void transpose_w1_f16(const float* __restrict__ W1, _Float16* __restrict__ w1t)
{
    const int k = blockIdx.x;
    const int n = threadIdx.x;
    w1t[(size_t)n * DIM + k] = (_Float16)W1[(size_t)k * DIM + n];
}

// ---------------------------------------------------------------------------
// Main kernel: persistent workgroups. Load W1^T into LDS once (TDM if
// available), then loop M-tiles: gather -> (e_p-e_q)^2 -> WMMA GEMM ->
// bias/ReLU -> xW2 reduction -> +b2.
// 256 threads = 8 wave32; wave w owns m-tile (w&1) and n-tiles (w>>1)+{0,4,8,12}.
// ---------------------------------------------------------------------------
__global__ __launch_bounds__(256)
void pair_mlp_wmma(const int*   __restrict__ p_idx,
                   const int*   __restrict__ q_idx,
                   const float* __restrict__ emb,
                   const _Float16* __restrict__ gW1t,   // f16 W1^T from pre-kernel
                   const float* __restrict__ b1,
                   const float* __restrict__ W2,
                   const float* __restrict__ b2,
                   float*       __restrict__ out,
                   int n_pairs)
{
    __shared__ _Float16 sW1t[DIM * SW];     // 135168 B (row stride 528B = 512+16 pad)
    __shared__ _Float16 sD[MTILE * SD];     //  16896 B
    __shared__ float    sAcc[MTILE];        //    128 B   (~149 KB total, < 320 KB)

    const int tid  = threadIdx.x;
    const int wave = tid >> 5;
    const int lane = tid & 31;

    // ---- stage W1^T (f16) into LDS once per (persistent) workgroup ----
#ifdef HAVE_TDM
    if (wave == 0) {
        const unsigned  lds_base = (unsigned)(unsigned long long)(void*)&sW1t[0];
        const unsigned long long ga = (unsigned long long)gW1t;

        // D# group 0: count=1 (valid user descriptor), lds_addr, global_addr, type=2
        u32x4 g0;
        g0.x = 1u;
        g0.y = lds_base;
        g0.z = (unsigned)ga;
        g0.w = (unsigned)((ga >> 32) & 0x01FFFFFFull) | 0x80000000u;  // addr[56:32] | type=2

        // D# group 1: data_size=1 (2B), pad_enable=1, pad_interval=6 (512B),
        // pad_amount=3 (16B) -> LDS row stride 528B == SW halves.
        // tensor_dim0=256, tensor_dim1=256, tile_dim0=256, tile_dim1=256,
        // tensor_dim0_stride=256 (contiguous rows in global).
        i32x8 g1;
        g1[0] = (int)((1u << 16) | (1u << 20) | (6u << 22) | (3u << 25)); // 0x07910000
        g1[1] = 0x01000000;   // tensor_dim0[15:0]=256 @ bits 63:48
        g1[2] = 0x01000000;   // tensor_dim1[15:0]=256 @ bits 95:80
        g1[3] = 0x01000000;   // tile_dim0=256        @ bits 127:112
        g1[4] = 0x00000100;   // tile_dim1=256        @ bits 143:128
        g1[5] = 0x00000100;   // tensor_dim0_stride=256
        g1[6] = 0;
        g1[7] = 0;

        i32x4 g2 = {0, 0, 0, 0};   // dims 2-4 unused (tile_dim2=0)
        i32x4 g3 = {0, 0, 0, 0};

#if __has_include(<hip/amd_detail/amd_gfx1250_TDM.h>)
        i32x8 g4 = {0, 0, 0, 0, 0, 0, 0, 0};
        __builtin_amdgcn_tensor_load_to_lds(g0, g1, g2, g3, g4, 0);      // 6-arg toolchain
#else
        __builtin_amdgcn_tensor_load_to_lds(g0, g1, g2, g3, 0);          // 5-arg ROCm 7.2
#endif
        __builtin_amdgcn_s_wait_tensorcnt(0);
    }
#else
    {
        #pragma unroll 4
        for (int i = tid; i < (DIM * DIM) / 8; i += 256) {
            const int n = i >> 5;
            const int c = i & 31;
            v8h w = *(const v8h*)(gW1t + (size_t)n * DIM + c * 8);
            *(v8h*)(&sW1t[n * SW + c * 8]) = w;
        }
    }
#endif
    __syncthreads();

    const int mt     = wave & 1;        // m-tile 0..1
    const int ntBase = wave >> 1;       // n-tile base 0..3 (tiles ntBase + 4*i)
    const int nl     = lane & 15;       // row-in-tile (A) / col-in-tile (B,C,D)
    const int hi     = lane >> 4;       // half-wave select
    const int arow   = (mt * 16 + nl) * SD;

    const int ntiles = (n_pairs + MTILE - 1) / MTILE;

    for (int tile = blockIdx.x; tile < ntiles; tile += gridDim.x) {
        const int m0 = tile * MTILE;

        if (tid < MTILE) sAcc[tid] = 0.0f;

        // ---- gather + squared difference -> f16 d tile in LDS ----
        {
            const int r  = tid >> 3;            // 0..31 : row of tile
            const int c0 = (tid & 7) * 32;      // 8 threads per 256-wide row
            int m = m0 + r;
            if (m >= n_pairs) m = n_pairs - 1;  // clamp (output guarded)
            const float* pr = emb + (size_t)p_idx[m] * DIM;
            const float* qr = emb + (size_t)q_idx[m] * DIM;
            #pragma unroll
            for (int c = 0; c < 32; c += 4) {
                v4f a = *(const v4f*)(pr + c0 + c);
                v4f b = *(const v4f*)(qr + c0 + c);
                v4f d = (a - b) * (a - b);
                *(v4h*)(&sD[r * SD + c0 + c]) = __builtin_convertvector(d, v4h);
            }
        }
        __syncthreads();

        // ---- WMMA GEMM: h = d @ W1, K = 256 in 8 steps of 32 ----
        v8f acc[4] = {};
        #pragma unroll
        for (int kt = 0; kt < 8; ++kt) {
            const int kb = kt * 32;
            // A fragment (16x32 f16): V0-3 = K{0..7 | 8..15}, V4-7 = K{16..23 | 24..31}
            V16 A;
            A.h[0] = *(const v8h*)(&sD[arow + kb + hi * 8]);
            A.h[1] = *(const v8h*)(&sD[arow + kb + 16 + hi * 8]);
            #pragma unroll
            for (int i = 0; i < 4; ++i) {
                const int nt   = ntBase + 4 * i;
                const int bofs = (nt * 16 + nl) * SW + kb + hi * 8;
                V16 B;
                B.h[0] = *(const v8h*)(&sW1t[bofs]);
                B.h[1] = *(const v8h*)(&sW1t[bofs + 16]);
                acc[i] = __builtin_amdgcn_wmma_f32_16x16x32_f16(
                             false, A.v, false, B.v, (short)0, acc[i], false, false);
            }
        }

        // ---- epilogue: +b1, ReLU, *W2[n], reduce over N into sAcc[m] ----
        #pragma unroll
        for (int i = 0; i < 4; ++i) {
            const int nt  = ntBase + 4 * i;
            const int n   = nt * 16 + nl;
            const float bias = b1[n];
            const float w2   = W2[n];
            #pragma unroll
            for (int v = 0; v < 8; ++v) {
                float hv = acc[i][v] + bias;
                hv = hv > 0.0f ? hv : 0.0f;
                const int mrow = mt * 16 + hi * 8 + v;   // C/D layout: M = hi*8 + vgpr
                __hip_atomic_fetch_add(&sAcc[mrow], hv * w2,
                                       __ATOMIC_RELAXED, __HIP_MEMORY_SCOPE_WORKGROUP);
            }
        }
        __syncthreads();

        if (tid < MTILE) {
            const int m = m0 + tid;
            if (m < n_pairs) out[m] = sAcc[tid] + b2[0];
        }
        __syncthreads();   // protect sAcc/sD reuse in next iteration
    }
}

// ---------------------------------------------------------------------------
extern "C" void kernel_launch(void* const* d_in, const int* in_sizes, int n_in,
                              void* d_out, int out_size, void* d_ws, size_t ws_size,
                              hipStream_t stream)
{
    const int*   p   = (const int*)  d_in[0];
    const int*   q   = (const int*)  d_in[1];
    const float* emb = (const float*)d_in[2];
    const float* W1  = (const float*)d_in[3];
    const float* b1  = (const float*)d_in[4];
    const float* W2  = (const float*)d_in[5];
    const float* b2  = (const float*)d_in[6];
    float*       out = (float*)d_out;
    const int n_pairs = in_sizes[0];

    _Float16* w1t = (_Float16*)d_ws;   // needs DIM*DIM*2 = 128 KB of scratch

    hipLaunchKernelGGL(transpose_w1_f16, dim3(DIM), dim3(DIM), 0, stream, W1, w1t);

    const int ntiles = (n_pairs + MTILE - 1) / MTILE;
    const int grid   = ntiles < NBLOCKS ? ntiles : NBLOCKS;
    hipLaunchKernelGGL(pair_mlp_wmma, dim3(grid), dim3(256), 0, stream,
                       p, q, emb, w1t, b1, W2, b2, out, n_pairs);
}
